// BQuantConv1d_toobig_10273561772174
// MI455X (gfx1250) — matrix operand order repaired and checked by reading, not compiled, not run on hardware
//
#include <hip/hip_runtime.h>
#include <hip/hip_bf16.h>

// ---------------------------------------------------------------------------
// BQuantConv1d == GEMM:  out[b,f] = bias[f] + sum_k x[b,k] * W[k,f]
//   W[k,f] = sum_q scale[q,f] * (+1/-1 from bit (7 - k%8) of binary[q, k/8, f])
// B=256, K=768, N=768, bits(q)=8, groups=96.
// Strategy: decode W once to f16 (N-major so WMMA B-fragments are contiguous),
// convert x to f16, then v_wmma_f32_16x16x32_f16 with f32 accumulation.
// ---------------------------------------------------------------------------

typedef _Float16 v16h __attribute__((ext_vector_type(16)));
typedef _Float16 v8h  __attribute__((ext_vector_type(8)));
typedef float    v8f  __attribute__((ext_vector_type(8)));

#define BQ_B   256
#define BQ_K   768
#define BQ_N   768
#define BQ_Q   8
#define BQ_G   96   // K/8

// ---- Kernel 1: x (f32) -> Xh (f16), row-major [B][K] ----------------------
__global__ void bq_xcvt_kernel(const float* __restrict__ x,
                               _Float16* __restrict__ Xh, int total) {
    int i = blockIdx.x * blockDim.x + threadIdx.x;
    if (i < total) Xh[i] = (_Float16)x[i];
}

// ---- Kernel 2: decode binary+scale -> Wt (f16), N-major [N][K] ------------
// grid = (3, 96), block = 256 : n = bx*256+tx (coalesced reads), g = by.
// Each thread emits 8 contiguous f16 (16B) at Wt[n*K + 8g].
__global__ void bq_decode_kernel(const int* __restrict__ binary,
                                 const float* __restrict__ scale,
                                 _Float16* __restrict__ Wt) {
    int n = blockIdx.x * blockDim.x + threadIdx.x;   // 0..767
    int g = blockIdx.y;                              // 0..95
    if (n >= BQ_N) return;
    float s[BQ_Q];
    int   b[BQ_Q];
#pragma unroll
    for (int q = 0; q < BQ_Q; ++q) {
        s[q] = scale[q * BQ_N + n];
        b[q] = binary[(q * BQ_G + g) * BQ_N + n];
    }
    v8h w;
#pragma unroll
    for (int p = 0; p < 8; ++p) {
        float acc = 0.0f;
#pragma unroll
        for (int q = 0; q < BQ_Q; ++q)
            acc += ((b[q] >> (7 - p)) & 1) ? s[q] : -s[q];
        w[p] = (_Float16)acc;
    }
    *(v8h*)(Wt + (size_t)n * BQ_K + g * 8) = w;      // 16B aligned
}

// ---- Kernel 3: WMMA GEMM --------------------------------------------------
// Block = 128 threads (4 waves, 2x2). Wave tile = 32x32 (4 accumulators).
// Grid = (N/64, B/64) = (12, 4).
union AFrag { v16h v; v8h h[2]; };

__global__ void __launch_bounds__(128)
bq_gemm_wmma_kernel(const _Float16* __restrict__ Xh,   // [B][K]
                    const _Float16* __restrict__ Wt,   // [N][K]
                    const float* __restrict__ bias,    // [N]
                    float* __restrict__ out) {         // [B][N]
    const int lane = threadIdx.x & 31;
    const int wave = threadIdx.x >> 5;                 // 0..3
    const int wm   = wave >> 1;                        // 0..1
    const int wn   = wave & 1;                         // 0..1
    const int mb   = blockIdx.y * 64 + wm * 32;
    const int nb   = blockIdx.x * 64 + wn * 32;
    const int l16  = lane & 15;
    const int hi   = lane >> 4;                        // 0 or 1

    v8f acc00 = {}, acc01 = {}, acc10 = {}, acc11 = {};

    // A: lane holds row m = mb(+16) + l16; k-halves = kb + 8*hi + {0..7, 16..23}
    const _Float16* xr0 = Xh + (size_t)(mb + l16) * BQ_K + 8 * hi;
    const _Float16* xr1 = xr0 + (size_t)16 * BQ_K;
    // B: lane holds col n = nb(+16) + l16; k-halves = kb + 8*hi + {0..7, 16..23}
    const _Float16* wr0 = Wt + (size_t)(nb + l16) * BQ_K + 8 * hi;
    const _Float16* wr1 = wr0 + (size_t)16 * BQ_K;

#pragma unroll 4
    for (int kb = 0; kb < BQ_K; kb += 32) {
        AFrag a0, a1, b0, b1;
        a0.h[0] = *(const v8h*)(xr0 + kb);
        a0.h[1] = *(const v8h*)(xr0 + kb + 16);
        a1.h[0] = *(const v8h*)(xr1 + kb);
        a1.h[1] = *(const v8h*)(xr1 + kb + 16);
        b0.h[0] = *(const v8h*)(wr0 + kb);
        b0.h[1] = *(const v8h*)(wr0 + kb + 16);
        b1.h[0] = *(const v8h*)(wr1 + kb);
        b1.h[1] = *(const v8h*)(wr1 + kb + 16);
        acc00 = __builtin_amdgcn_wmma_f32_16x16x32_f16(false, a0.v, false, b0.v,
                                                       (short)0, acc00, false, false);
        acc01 = __builtin_amdgcn_wmma_f32_16x16x32_f16(false, a0.v, false, b1.v,
                                                       (short)0, acc01, false, false);
        acc10 = __builtin_amdgcn_wmma_f32_16x16x32_f16(false, a1.v, false, b0.v,
                                                       (short)0, acc10, false, false);
        acc11 = __builtin_amdgcn_wmma_f32_16x16x32_f16(false, a1.v, false, b1.v,
                                                       (short)0, acc11, false, false);
    }

    // C/D layout: VGPR r -> row (r + 8*hi), col l16.
    const float bv0 = bias[nb + l16];
    const float bv1 = bias[nb + 16 + l16];
#pragma unroll
    for (int r = 0; r < 8; ++r) {
        int row0 = mb + r + 8 * hi;
        int row1 = row0 + 16;
        out[(size_t)row0 * BQ_N + nb + l16]      = acc00[r] + bv0;
        out[(size_t)row0 * BQ_N + nb + 16 + l16] = acc01[r] + bv1;
        out[(size_t)row1 * BQ_N + nb + l16]      = acc10[r] + bv0;
        out[(size_t)row1 * BQ_N + nb + 16 + l16] = acc11[r] + bv1;
    }
}

// ---- Fallback (ws too small): direct f32 evaluation -----------------------
__global__ void bq_fallback_kernel(const float* __restrict__ x,
                                   const float* __restrict__ scale,
                                   const int* __restrict__ binary,
                                   const float* __restrict__ bias,
                                   float* __restrict__ out) {
    int idx = blockIdx.x * blockDim.x + threadIdx.x;
    if (idx >= BQ_B * BQ_N) return;
    int f = idx % BQ_N, b = idx / BQ_N;
    float acc = bias[f];
    for (int q = 0; q < BQ_Q; ++q) {
        float sq = scale[q * BQ_N + f];
        float t = 0.0f;
        for (int g = 0; g < BQ_G; ++g) {
            int byte = binary[(q * BQ_G + g) * BQ_N + f];
#pragma unroll
            for (int p = 0; p < 8; ++p) {
                float xv = x[(size_t)b * BQ_K + g * 8 + p];
                t += ((byte >> (7 - p)) & 1) ? xv : -xv;
            }
        }
        acc += sq * t;
    }
    out[idx] = acc;
}

extern "C" void kernel_launch(void* const* d_in, const int* in_sizes, int n_in,
                              void* d_out, int out_size, void* d_ws, size_t ws_size,
                              hipStream_t stream) {
    const float* x      = (const float*)d_in[0];   // (4,64,768) f32
    const float* scale  = (const float*)d_in[1];   // (1,8,768)  f32
    const int*   binary = (const int*)  d_in[2];   // (1,8,96,768) i32
    const float* bias   = (const float*)d_in[3];   // (768,)     f32
    float*       out    = (float*)d_out;           // (4,64,768) f32

    const size_t xh_bytes = (size_t)BQ_B * BQ_K * sizeof(_Float16); // 384 KB
    const size_t wt_bytes = (size_t)BQ_N * BQ_K * sizeof(_Float16); // 1.125 MB

    if (ws_size < xh_bytes + wt_bytes) {
        int total = BQ_B * BQ_N;
        bq_fallback_kernel<<<(total + 255) / 256, 256, 0, stream>>>(x, scale, binary, bias, out);
        return;
    }

    _Float16* Xh = (_Float16*)d_ws;
    _Float16* Wt = (_Float16*)((char*)d_ws + xh_bytes);

    int xtotal = BQ_B * BQ_K;
    bq_xcvt_kernel<<<(xtotal + 255) / 256, 256, 0, stream>>>(x, Xh, xtotal);
    bq_decode_kernel<<<dim3(BQ_N / 256, BQ_G), 256, 0, stream>>>(binary, scale, Wt);
    bq_gemm_wmma_kernel<<<dim3(BQ_N / 64, BQ_B / 64), 128, 0, stream>>>(Xh, Wt, bias, out);
}
